// DVGGA_75617194213710
// MI455X (gfx1250) — matrix operand order repaired and checked by reading, not compiled, not run on hardware
//
#include <hip/hip_runtime.h>
#include <hip/hip_bf16.h>
#include <cstdint>

// ---------------------------------------------------------------------------
// CDNA5 (gfx1250) WMMA helpers: v_wmma_f32_16x16x32_bf16, wave32 layouts.
// ---------------------------------------------------------------------------
typedef __bf16 bf16_t;
typedef __attribute__((ext_vector_type(16))) __bf16 v16bf;
typedef __attribute__((ext_vector_type(8)))  float  v8f;

#define WMMA_BF16(a, b, c) \
    __builtin_amdgcn_wmma_f32_16x16x32_bf16(false, (a), false, (b), (short)0, (c), false, false)

// A fragment: 16x32 (MxK), 16-bit. Lane L holds row M=L%16; element j -> K=(L/16)*16+j
// (contiguous k order, matching the swizzled B fragments).
// Rows are contiguous in the source -> lowers to contiguous ds_load_b128s.
template <typename T>
__device__ inline v16bf load_A_frag(const T* src, int lda, int row0, int k0) {
    const int lane = threadIdx.x & 31;
    const int m  = row0 + (lane & 15);
    const int kb = k0 + ((lane >> 4) << 4);
    v16bf a;
#pragma unroll
    for (int j = 0; j < 8; ++j) {
        a[j]     = (bf16_t)(float)src[m * lda + kb + j];
        a[j + 8] = (bf16_t)(float)src[m * lda + kb + 8 + j];
    }
    return a;
}

// B fragment from pre-swizzled LDS storage laid out as [lane][j] (32 lanes x 16 bf16):
// element (k,n) of the 32x16 fragment lives at lane=(k>>4)*16 + n, j = k&15.
// One aligned 32-byte contiguous read per lane -> 2x ds_load_b128, no packing VALU.
__device__ inline v16bf load_B_frag_sw(const bf16_t* fragbase) {
    return *(const v16bf*)(fragbase + ((threadIdx.x & 31) << 4));
}

// ---------------------------------------------------------------------------
// Problem constants
// ---------------------------------------------------------------------------
#define GG 4096
#define NN 64
#define EE 256
#define FF 64
#define DD 128
#define HH1 16
#define EPN 32768

// ---------------------------------------------------------------------------
// Kernel 1: fused per-graph pipeline (1 block = 1 graph, 256 thr = 8 waves)
// ---------------------------------------------------------------------------
__global__ void __launch_bounds__(256) sage_graph_kernel(
    const float* __restrict__ x, const int* __restrict__ edge_index,
    const float* __restrict__ edge_weight,
    const float* __restrict__ W_gcn1, const float* __restrict__ b_gcn1,
    const float* __restrict__ W_fc1,  const float* __restrict__ b_fc1,
    const float* __restrict__ W_fc2,  const float* __restrict__ b_fc2,
    float* __restrict__ g_embs, float* __restrict__ pens)
{
    __shared__ __align__(32) char s_raw[24 * 1024];   // phase-overlaid pool
    __shared__ float s_xw[NN * DD];                   // xw -> h   (32 KB)
    __shared__ float s_deg[NN], s_dinv[NN], s_degL[NN], s_dinvL[NN];
    __shared__ float s_S[NN * 2], s_LS[NN * 2];
    __shared__ float s_red[256];

    const int g    = blockIdx.x;
    const int tid  = threadIdx.x;
    const int wave = tid >> 5;

    bf16_t* xbf    = (bf16_t*)s_raw;                // [64*64]  linear   (8 KB)
    bf16_t* wbf_sw = (bf16_t*)(s_raw + 8192);       // [16 frags * 512]  (16 KB)

    const float* xg = x + (size_t)g * NN * FF;
    for (int i = tid; i < NN * FF; i += 256) xbf[i] = (bf16_t)xg[i];
    // Stage W_gcn1 [64,128] directly in B-fragment layout:
    // frag f = (kpanel<<3) | ntile; within frag: [lane][j].
#pragma unroll
    for (int i = 0; i < 32; ++i) {
        int idx  = tid * 32 + i;          // contiguous dst -> vector ds stores
        int f    = idx >> 9;              // 0..15
        int r    = idx & 511;
        int lane = r >> 4, j = r & 15;
        int k    = ((f >> 3) << 5) + ((lane >> 4) << 4) + j;
        int n    = ((f & 7) << 4) + (lane & 15);
        wbf_sw[idx] = (bf16_t)W_gcn1[k * DD + n];
    }
    if (tid < NN) { s_deg[tid] = 0.f; s_degL[tid] = 0.f; }

    const int*  eig = edge_index + (size_t)g * 2 * EE;
    const int   es  = eig[tid];
    const int   ed  = eig[EE + tid];
    const float ew  = edge_weight[(size_t)g * EE + tid];
    __syncthreads();

    atomicAdd(&s_deg[ed],  ew);   // GCN degree (dst)
    atomicAdd(&s_degL[es], ew);   // Laplacian degree (src)

    // GEMM1: xw[64,128] = x[64,64] @ W_gcn1[64,128];  wave w owns N-tile w.
    {
        const int nt = wave;
#pragma unroll
        for (int mt = 0; mt < 4; ++mt) {
            v8f acc = {};
#pragma unroll
            for (int kp = 0; kp < 2; ++kp) {
                v16bf a = load_A_frag(xbf, FF, mt * 16, kp * 32);
                v16bf b = load_B_frag_sw(wbf_sw + (((kp << 3) | nt) << 9));
                acc = WMMA_BF16(a, b, acc);
            }
            const int lane = tid & 31, nn = lane & 15, mb = (lane >> 4) * 8;
#pragma unroll
            for (int r = 0; r < 8; ++r)
                s_xw[(mt * 16 + mb + r) * DD + nt * 16 + nn] = acc[r];
        }
    }
    __syncthreads();

    if (tid < NN) {
        s_dinv[tid] = rsqrtf(s_deg[tid] + 1.0f);          // +1 self loop
        float dl    = s_degL[tid];
        s_dinvL[tid] = (dl > 0.f) ? rsqrtf(fmaxf(dl, 1e-30f)) : 0.f;
    }
    __syncthreads();
    const float enorm = s_dinv[es] * ew * s_dinv[ed];

    // GCN aggregation, two 64-column halves to fit LDS (agg reuses s_raw).
    float* agg = (float*)s_raw;                  // [64*64] (16 KB)
    for (int half = 0; half < 2; ++half) {
        for (int i = tid; i < NN * 64; i += 256) agg[i] = 0.f;
        __syncthreads();
        for (int c = 0; c < 64; ++c)
            atomicAdd(&agg[ed * 64 + c], s_xw[es * DD + half * 64 + c] * enorm);
        __syncthreads();
        for (int i = tid; i < NN * 64; i += 256) {
            int n = i >> 6, c = i & 63;
            float self = s_dinv[n] * s_dinv[n];
            s_xw[n * DD + half * 64 + c] =
                agg[i] + s_xw[n * DD + half * 64 + c] * self + b_gcn1[half * 64 + c];
        }
        __syncthreads();
    }
    // s_xw now holds h[64,128]

    bf16_t* wfc1sw = (bf16_t*)(s_raw + 16384);   // [4 frags * 512] (4 KB)
    float*  a1     = (float*)(s_raw + 20480);    // [64*16]  (4 KB)
    // Stage W_fc1 [128,16] in B-fragment layout: frag f = k-panel 0..3.
#pragma unroll
    for (int i = 0; i < 8; ++i) {
        int idx  = tid * 8 + i;           // 0..2047
        int f    = idx >> 9;
        int r    = idx & 511;
        int lane = r >> 4, j = r & 15;
        int k    = (f << 5) + ((lane >> 4) << 4) + j;
        int n    = lane & 15;
        wfc1sw[idx] = (bf16_t)W_fc1[k * HH1 + n];
    }
    __syncthreads();

    // GEMM2: a1[64,16] = tanh(h @ W_fc1 + b_fc1); waves 0..3 own M-tiles.
    if (wave < 4) {
        const int mt = wave;
        v8f acc = {};
#pragma unroll
        for (int kp = 0; kp < 4; ++kp) {
            v16bf a = load_A_frag(s_xw, DD, mt * 16, kp * 32);
            v16bf b = load_B_frag_sw(wfc1sw + (kp << 9));
            acc = WMMA_BF16(a, b, acc);
        }
        const int lane = tid & 31, nn = lane & 15, mb = (lane >> 4) * 8;
        float bv = b_fc1[nn];
#pragma unroll
        for (int r = 0; r < 8; ++r)
            a1[(mt * 16 + mb + r) * HH1 + nn] = tanhf(acc[r] + bv);
    }
    __syncthreads();

    // fc2 + softmax (2 classes)
    if (tid < NN) {
        float l0 = b_fc2[0], l1 = b_fc2[1];
#pragma unroll
        for (int j = 0; j < HH1; ++j) {
            float v = a1[tid * HH1 + j];
            l0 += v * W_fc2[j * 2 + 0];
            l1 += v * W_fc2[j * 2 + 1];
        }
        float m = fmaxf(l0, l1);
        float e0 = expf(l0 - m), e1 = expf(l1 - m);
        float inv = 1.0f / (e0 + e1);
        float s0 = e0 * inv, s1 = e1 * inv;
        s_S[tid * 2 + 0] = s0;  s_S[tid * 2 + 1] = s1;
        s_LS[tid * 2 + 0] = s0; s_LS[tid * 2 + 1] = s1;   // identity part of L@S
    }
    __syncthreads();

    // L@S scatter: (L@S)[src] += lw * S[dst]
    {
        float lw = -s_dinvL[es] * ew * s_dinvL[ed];
        atomicAdd(&s_LS[es * 2 + 0], lw * s_S[ed * 2 + 0]);
        atomicAdd(&s_LS[es * 2 + 1], lw * s_S[ed * 2 + 1]);
    }
    __syncthreads();

    if (tid < NN) {
        s_red[tid * 4 + 0] = s_S[tid * 2 + 0] * s_LS[tid * 2 + 0];
        s_red[tid * 4 + 1] = s_S[tid * 2 + 0] * s_LS[tid * 2 + 1];
        s_red[tid * 4 + 2] = s_S[tid * 2 + 1] * s_LS[tid * 2 + 0];
        s_red[tid * 4 + 3] = s_S[tid * 2 + 1] * s_LS[tid * 2 + 1];
    }
    __syncthreads();
    if (tid == 0) {
        float a00 = 0, a01 = 0, a10 = 0, a11 = 0;
        for (int n = 0; n < NN; ++n) {
            a00 += s_red[n * 4 + 0]; a01 += s_red[n * 4 + 1];
            a10 += s_red[n * 4 + 2]; a11 += s_red[n * 4 + 3];
        }
        float r0 = fmaxf(fabsf(a00) + fabsf(a01), 1e-12f);
        float r1 = fmaxf(fabsf(a10) + fabsf(a11), 1e-12f);
        float d0 = a00 / r0 - 1.0f, d1 = a11 / r1 - 1.0f;
        pens[g] = 0.5f * (d0 * d0 + d1 * d1);
    }
    // g_emb[c] = 0.5 * sum_n (S[n,0]+S[n,1]) * h[n,c]
    if (tid < DD) {
        float acc = 0.f;
        for (int n = 0; n < NN; ++n)
            acc += 0.5f * (s_S[n * 2] + s_S[n * 2 + 1]) * s_xw[n * DD + tid];
        g_embs[(size_t)g * DD + tid] = acc;
    }
}

// ---------------------------------------------------------------------------
// Kernel 2: generic WMMA GEMM  C[M,N] = A[M,K] @ B[K,N] (+bias), 64x64 tiles.
// A panel staged f32 via CDNA5 async global->LDS DMA (ASYNCcnt); B panel
// staged bf16 pre-swizzled into fragment layout. 8 waves, 2 C tiles each.
// ---------------------------------------------------------------------------
__global__ void __launch_bounds__(256) wmma_gemm_bias(
    const float* __restrict__ A, const float* __restrict__ B,
    const float* __restrict__ bias, float* __restrict__ C,
    int M, int N, int K)
{
    __shared__ __align__(16) float  sA[64 * 32];     // raw f32, async-copied (8 KB)
    __shared__ __align__(32) bf16_t sBsw[4 * 512];   // 4 fragment blocks     (4 KB)
    const int bn = blockIdx.x, bm = blockIdx.y;
    const int tid  = threadIdx.x;
    const int wave = tid >> 5;
    const int mt   = wave >> 1;
    const int nt0  = (wave & 1) * 2;

    v8f acc0 = {}; v8f acc1 = {};
    for (int k0 = 0; k0 < K; k0 += 32) {
        // --- A panel 64x32 f32: two async b128 copies per thread ---
        {
            int idx0 = tid * 4;           // 4 floats = 16 B
            int idx1 = idx0 + 1024;
            const float* g0 = &A[(size_t)(bm * 64 + (idx0 >> 5)) * K + k0 + (idx0 & 31)];
            const float* g1 = &A[(size_t)(bm * 64 + (idx1 >> 5)) * K + k0 + (idx1 & 31)];
            unsigned l0 = (unsigned)(uintptr_t)(&sA[idx0]);
            unsigned l1 = (unsigned)(uintptr_t)(&sA[idx1]);
            asm volatile("global_load_async_to_lds_b128 %0, %1, off"
                         :: "v"(l0), "v"(g0) : "memory");
            asm volatile("global_load_async_to_lds_b128 %0, %1, off"
                         :: "v"(l1), "v"(g1) : "memory");
        }
        if (k0 + 32 < K)  // next B panel -> global_prefetch_b8
            __builtin_prefetch(&B[(size_t)(k0 + 32 + (tid & 31)) * N + bn * 64], 0, 1);
        // --- B panel 32x64 -> bf16 in fragment layout (contiguous dst) ---
#pragma unroll
        for (int i = 0; i < 8; ++i) {
            int idx  = tid * 8 + i;       // 0..2047
            int f    = idx >> 9;          // n-tile 0..3
            int r    = idx & 511;
            int lane = r >> 4, j = r & 15;
            int k    = ((lane >> 4) << 4) + j;
            int n    = (f << 4) + (lane & 15);
            sBsw[idx] = (bf16_t)B[(size_t)(k0 + k) * N + bn * 64 + n];
        }
        asm volatile("s_wait_asynccnt 0x0" ::: "memory");
        __syncthreads();

        v16bf a  = load_A_frag(sA, 32, mt * 16, 0);       // f32 -> bf16 on load
        v16bf b0 = load_B_frag_sw(sBsw + (nt0 << 9));
        v16bf b1 = load_B_frag_sw(sBsw + ((nt0 + 1) << 9));
        acc0 = WMMA_BF16(a, b0, acc0);
        acc1 = WMMA_BF16(a, b1, acc1);
        __syncthreads();
    }
    const int lane = tid & 31, nn = lane & 15, mb = (lane >> 4) * 8;
#pragma unroll
    for (int t = 0; t < 2; ++t) {
        v8f acc = t ? acc1 : acc0;
        int col = bn * 64 + (nt0 + t) * 16 + nn;
        float bv = bias ? bias[col] : 0.0f;
#pragma unroll
        for (int r = 0; r < 8; ++r) {
            int row = bm * 64 + mt * 16 + mb + r;
            C[(size_t)row * N + col] = acc[r] + bv;
        }
    }
}

// ---------------------------------------------------------------------------
// Graph-level GCN support kernels
// ---------------------------------------------------------------------------
__global__ void __launch_bounds__(256) deg_kernel(const int* __restrict__ pe,
                                                  float* __restrict__ deg) {
    int e = blockIdx.x * 256 + threadIdx.x;
    if (e < EPN) atomicAdd(&deg[pe[EPN + e]], 1.0f);
}
__global__ void __launch_bounds__(256) dinv_kernel(float* __restrict__ deg) {
    int i = blockIdx.x * 256 + threadIdx.x;
    if (i < GG) deg[i] = rsqrtf(deg[i] + 1.0f);   // in-place deg -> dinv
}
// agg[dst] += X[src]*dinv[src]*dinv[dst]; grid = EPN, block = dim
__global__ void scatter_agg(const int* __restrict__ ei, const float* __restrict__ dinv,
                            const float* __restrict__ X, float* __restrict__ AGG, int dim) {
    int e = blockIdx.x, c = threadIdx.x;
    int s = ei[e], d = ei[EPN + e];
    float nrm = dinv[s] * dinv[d];
    atomicAdd(&AGG[(size_t)d * dim + c], X[(size_t)s * dim + c] * nrm);
}
// dual scatter for mu/ls (shared edges), block = 128
__global__ void __launch_bounds__(128) scatter_agg2(
    const int* __restrict__ ei, const float* __restrict__ dinv,
    const float* __restrict__ Xa, float* __restrict__ Aa,
    const float* __restrict__ Xb, float* __restrict__ Ab) {
    int e = blockIdx.x, c = threadIdx.x;
    int s = ei[e], d = ei[EPN + e];
    float nrm = dinv[s] * dinv[d];
    atomicAdd(&Aa[(size_t)d * DD + c], Xa[(size_t)s * DD + c] * nrm);
    atomicAdd(&Ab[(size_t)d * DD + c], Xb[(size_t)s * DD + c] * nrm);
}
__global__ void __launch_bounds__(256) combine_relu(
    const float* __restrict__ XW, const float* __restrict__ AGG,
    const float* __restrict__ dinv, const float* __restrict__ bias,
    float* __restrict__ OUT, int dim) {
    int i = blockIdx.x, c = threadIdx.x;
    float di2 = dinv[i] * dinv[i];
    OUT[(size_t)i * dim + c] =
        fmaxf(AGG[(size_t)i * dim + c] + XW[(size_t)i * dim + c] * di2 + bias[c], 0.f);
}
// mu / logstd / z / zc / per-row KL partial; grid = G, block = 128
__global__ void __launch_bounds__(128) vgae_finish(
    const float* __restrict__ XWmu, const float* __restrict__ AGGmu,
    const float* __restrict__ XWls, const float* __restrict__ AGGls,
    const float* __restrict__ dinv, const float* __restrict__ b_mu,
    const float* __restrict__ b_ls, const float* __restrict__ eps,
    const float* __restrict__ emb, float* __restrict__ zc,
    float* __restrict__ klpart) {
    int i = blockIdx.x, d = threadIdx.x;
    float di2 = dinv[i] * dinv[i];
    float mu = AGGmu[(size_t)i * DD + d] + XWmu[(size_t)i * DD + d] * di2 + b_mu[d];
    float ls = AGGls[(size_t)i * DD + d] + XWls[(size_t)i * DD + d] * di2 + b_ls[d];
    ls = fminf(ls, 10.0f);
    float z = mu + eps[(size_t)i * DD + d] * expf(ls);
    zc[(size_t)i * 256 + d]       = emb[(size_t)i * DD + d];
    zc[(size_t)i * 256 + DD + d]  = z;
    float kle = 1.0f + 2.0f * ls - mu * mu - expf(2.0f * ls);
    __shared__ float red[128];
    red[d] = kle; __syncthreads();
    for (int s = 64; s > 0; s >>= 1) { if (d < s) red[d] += red[d + s]; __syncthreads(); }
    if (d == 0) klpart[i] = red[0];
}

// ---------------------------------------------------------------------------
// Link prediction: one wave per edge (32 lanes x 8 dims = 256), shuffle reduce.
// ---------------------------------------------------------------------------
__global__ void __launch_bounds__(256) pred_kernel(
    const float* __restrict__ ZL1, const float* __restrict__ ZL2,
    const int* __restrict__ ei, float* __restrict__ pred,
    float* __restrict__ logpart, int isneg) {
    const int tid = threadIdx.x, wave = tid >> 5, lane = tid & 31;
    const int e = blockIdx.x * 8 + wave;
    float logv = 0.f;
    if (e < EPN) {
        int a = ei[e], b = ei[EPN + e];
        float s = 0.f;
#pragma unroll
        for (int j = 0; j < 8; ++j)
            s += ZL1[(size_t)a * 256 + lane * 8 + j] * ZL2[(size_t)b * 256 + lane * 8 + j];
        for (int off = 16; off > 0; off >>= 1) s += __shfl_xor(s, off, 32);
        float p = 1.0f / (1.0f + expf(-s));
        if (lane == 0) pred[e] = p;
        logv = isneg ? logf(1.0f - p + 1e-15f) : logf(p + 1e-15f);
    }
    __shared__ float red[8];
    if (lane == 0) red[wave] = (e < EPN) ? logv : 0.f;
    __syncthreads();
    if (tid == 0) {
        float t = 0.f;
        for (int w = 0; w < 8; ++w) t += red[w];
        logpart[blockIdx.x] = t;
    }
}

// Final deterministic reduction of the four 4096-length partial arrays.
__global__ void __launch_bounds__(256) final_kernel(
    const float* __restrict__ pospart, const float* __restrict__ negpart,
    const float* __restrict__ klpart, const float* __restrict__ pens,
    float* __restrict__ out) {
    int tid = threadIdx.x;
    float sp = 0, sn = 0, sk = 0, se = 0;
    for (int i = tid; i < GG; i += 256) {
        sp += pospart[i]; sn += negpart[i]; sk += klpart[i]; se += pens[i];
    }
    __shared__ float rp[256], rn[256], rk[256], re[256];
    rp[tid] = sp; rn[tid] = sn; rk[tid] = sk; re[tid] = se;
    __syncthreads();
    for (int s = 128; s > 0; s >>= 1) {
        if (tid < s) { rp[tid] += rp[tid+s]; rn[tid] += rn[tid+s];
                       rk[tid] += rk[tid+s]; re[tid] += re[tid+s]; }
        __syncthreads();
    }
    if (tid == 0) {
        float rec = -(rp[0] / (float)EPN) - (rn[0] / (float)EPN);
        float kl  = -0.5f * (rk[0] / (float)GG);
        out[0] = rec + kl / (float)GG;      // pre_loss
        out[1] = re[0] / (float)GG;         // positive_penalty
    }
}

// ---------------------------------------------------------------------------
// Launcher
// ---------------------------------------------------------------------------
extern "C" void kernel_launch(void* const* d_in, const int* in_sizes, int n_in,
                              void* d_out, int out_size, void* d_ws, size_t ws_size,
                              hipStream_t stream) {
    (void)in_sizes; (void)n_in; (void)out_size; (void)ws_size;
    const float* x        = (const float*)d_in[0];
    const int*   ei       = (const int*)  d_in[1];
    const float* ew       = (const float*)d_in[2];
    const int*   pos_e    = (const int*)  d_in[3];
    const int*   neg_e    = (const int*)  d_in[4];
    const float* eps      = (const float*)d_in[5];
    const float* W_gcn1   = (const float*)d_in[6];
    const float* b_gcn1   = (const float*)d_in[7];
    const float* W_fc1    = (const float*)d_in[8];
    const float* b_fc1    = (const float*)d_in[9];
    const float* W_fc2    = (const float*)d_in[10];
    const float* b_fc2    = (const float*)d_in[11];
    const float* W_c1     = (const float*)d_in[12];
    const float* b_c1     = (const float*)d_in[13];
    const float* W_mu     = (const float*)d_in[14];
    const float* b_mu     = (const float*)d_in[15];
    const float* W_ls     = (const float*)d_in[16];
    const float* b_ls     = (const float*)d_in[17];
    const float* emb      = (const float*)d_in[18];
    const float* W_lin1   = (const float*)d_in[19];
    const float* b_lin1   = (const float*)d_in[20];
    const float* W_lin2   = (const float*)d_in[21];
    const float* b_lin2   = (const float*)d_in[22];
    float* out = (float*)d_out;
    float* ws  = (float*)d_ws;

    // workspace layout (float offsets)
    constexpr size_t O_GEMB  = 0;
    constexpr size_t O_PENS  = O_GEMB  + (size_t)GG * DD;
    constexpr size_t O_DINV  = O_PENS  + GG;            // deg -> dinv in place
    constexpr size_t O_XW1   = O_DINV  + GG;
    constexpr size_t O_AGG1  = O_XW1   + (size_t)GG * 256;
    constexpr size_t O_H1    = O_AGG1  + (size_t)GG * 256;
    constexpr size_t O_XWMU  = O_H1    + (size_t)GG * 256;
    constexpr size_t O_XWLS  = O_XWMU  + (size_t)GG * DD;
    constexpr size_t O_AGGMU = O_XWLS  + (size_t)GG * DD;
    constexpr size_t O_AGGLS = O_AGGMU + (size_t)GG * DD;
    constexpr size_t O_ZC    = O_AGGLS + (size_t)GG * DD;
    constexpr size_t O_ZL1   = O_ZC    + (size_t)GG * 256;
    constexpr size_t O_ZL2   = O_ZL1   + (size_t)GG * 256;
    constexpr size_t O_PPART = O_ZL2   + (size_t)GG * 256;
    constexpr size_t O_NPART = O_PPART + GG;
    constexpr size_t O_KPART = O_NPART + GG;

    float* gembs = ws + O_GEMB;  float* pens  = ws + O_PENS;
    float* dinv  = ws + O_DINV;  float* xw1   = ws + O_XW1;
    float* agg1  = ws + O_AGG1;  float* h1    = ws + O_H1;
    float* xwmu  = ws + O_XWMU;  float* xwls  = ws + O_XWLS;
    float* aggmu = ws + O_AGGMU; float* aggls = ws + O_AGGLS;
    float* zc    = ws + O_ZC;    float* zl1   = ws + O_ZL1;
    float* zl2   = ws + O_ZL2;
    float* ppart = ws + O_PPART; float* npart = ws + O_NPART;
    float* kpart = ws + O_KPART;

    // zero the accumulation buffers (graph-capturable stream memsets)
    hipMemsetAsync(dinv,  0, (size_t)GG * sizeof(float), stream);
    hipMemsetAsync(agg1,  0, (size_t)GG * 256 * sizeof(float), stream);
    hipMemsetAsync(aggmu, 0, (size_t)GG * DD * sizeof(float), stream);
    hipMemsetAsync(aggls, 0, (size_t)GG * DD * sizeof(float), stream);

    // 1) fused per-graph pipeline (4096 blocks)
    sage_graph_kernel<<<GG, 256, 0, stream>>>(x, ei, ew, W_gcn1, b_gcn1,
                                              W_fc1, b_fc1, W_fc2, b_fc2,
                                              gembs, pens);
    // 2) graph-level GCN degree / dinv
    deg_kernel<<<EPN / 256, 256, 0, stream>>>(pos_e, dinv);
    dinv_kernel<<<GG / 256, 256, 0, stream>>>(dinv);
    // 3) h = relu(gcn(g_embs, W_c1, b_c1))
    wmma_gemm_bias<<<dim3(256 / 64, GG / 64), 256, 0, stream>>>(gembs, W_c1, nullptr, xw1, GG, 256, DD);
    scatter_agg<<<EPN, 256, 0, stream>>>(pos_e, dinv, xw1, agg1, 256);
    combine_relu<<<GG, 256, 0, stream>>>(xw1, agg1, dinv, b_c1, h1, 256);
    // 4) mu / logstd GEMMs + scatter
    wmma_gemm_bias<<<dim3(DD / 64, GG / 64), 256, 0, stream>>>(h1, W_mu, nullptr, xwmu, GG, DD, 256);
    wmma_gemm_bias<<<dim3(DD / 64, GG / 64), 256, 0, stream>>>(h1, W_ls, nullptr, xwls, GG, DD, 256);
    scatter_agg2<<<EPN, 128, 0, stream>>>(pos_e, dinv, xwmu, aggmu, xwls, aggls);
    // 5) mu, logstd, z, zc, KL partials
    vgae_finish<<<GG, 128, 0, stream>>>(xwmu, aggmu, xwls, aggls, dinv,
                                        b_mu, b_ls, eps, emb, zc, kpart);
    // 6) hoisted link projections: ZL = zc @ W_lin + b  (instead of per-edge matvecs)
    wmma_gemm_bias<<<dim3(256 / 64, GG / 64), 256, 0, stream>>>(zc, W_lin1, b_lin1, zl1, GG, 256, 256);
    wmma_gemm_bias<<<dim3(256 / 64, GG / 64), 256, 0, stream>>>(zc, W_lin2, b_lin2, zl2, GG, 256, 256);
    // 7) predictions + log partials
    pred_kernel<<<EPN / 8, 256, 0, stream>>>(zl1, zl2, pos_e, out + 2,        ppart, 0);
    pred_kernel<<<EPN / 8, 256, 0, stream>>>(zl1, zl2, neg_e, out + 2 + EPN,  npart, 1);
    // 8) scalars
    final_kernel<<<1, 256, 0, stream>>>(ppart, npart, kpart, pens, out);
}
